// LearnableMultiheadSelfAttention_47098611368152
// MI455X (gfx1250) — matrix operand order repaired and checked by reading, compiled, not run on hardware
//
#include <hip/hip_runtime.h>

#define L_  2048
#define B_  4
#define D_  1024
#define H_  16
#define DH_ 64
#define LB_ (L_ * B_)    // 8192 rows
#define HD_ (H_ * DH_)   // 1024 cols

#define OP_QUQV 0
#define OP_KV   1
#define OP_R    2
#define OP_OUT  3

typedef __bf16 v16bf __attribute__((ext_vector_type(16)));
typedef float  v8f   __attribute__((ext_vector_type(8)));

struct U4 { unsigned x, y, z, w; };

union Frag16 {
  v16bf v;
  U4 q[2];
  unsigned short s[16];
};

#define WMMA_BF16(a, b, c) \
  __builtin_amdgcn_wmma_f32_16x16x32_bf16(false, (a), false, (b), (short)0, (c), false, false)

__device__ __forceinline__ unsigned short f2bf(float f) {
  union { float f; unsigned u; } c; c.f = f;
  unsigned u = c.u;
  unsigned r = (u + 0x7FFFu + ((u >> 16) & 1u)) >> 16;
  if ((u & 0x7F800000u) == 0x7F800000u) r = u >> 16;  // Inf/NaN passthrough
  return (unsigned short)r;
}

__device__ __forceinline__ v8f v8f_zero() {
  v8f z;
#pragma unroll
  for (int e = 0; e < 8; ++e) z[e] = 0.f;
  return z;
}

__global__ void cvt_f32_bf16(const float* __restrict__ src,
                             unsigned short* __restrict__ dst, int n) {
  int i = blockIdx.x * blockDim.x + threadIdx.x;
  if (i < n) dst[i] = f2bf(src[i]);
}

// ---------------------------------------------------------------------------
// bf16 WMMA GEMM: C[M,N] = A[M,K] @ B[K,N], fp32 accumulate.
// Workgroup = 128 threads (4 waves). 128x64 C tile; wave = 32x64 strip
// (2 M-tiles -> 8 WMMAs per K-step per wave, B LDS traffic amortized 2x).
// B tile staged transposed in LDS (vector global loads + b16 scatter) so
// B-fragments are contiguous 16B LDS loads. Epilogue scatters per op.
// ---------------------------------------------------------------------------
__global__ __launch_bounds__(128)
void gemm_wmma(const unsigned short* __restrict__ A,
               const unsigned short* __restrict__ Bw,
               int M, int N, int K, int op,
               const float* __restrict__ bias_u,
               const float* __restrict__ bias_v,
               unsigned short* __restrict__ o_qu,
               unsigned short* __restrict__ o_qv,
               unsigned short* __restrict__ o_k,
               unsigned short* __restrict__ o_vt,
               unsigned short* __restrict__ o_r,
               float* __restrict__ o_f32) {
  __shared__ unsigned short Bt[64][40];  // [n][k], padded rows (80B, 16B aligned)
  const int tid  = threadIdx.x;
  const int wave = tid >> 5, lane = tid & 31;
  const int hi   = lane >> 4, ln = lane & 15;
  const int bm = blockIdx.x * 128, bn = blockIdx.y * 64;

  v8f acc[2][4];
#pragma unroll
  for (int mt = 0; mt < 2; ++mt)
#pragma unroll
    for (int nt = 0; nt < 4; ++nt) acc[mt][nt] = v8f_zero();

  for (int k0 = 0; k0 < K; k0 += 32) {
    __syncthreads();
    {
      // stage B[32x64] transposed: 2x global_load_b128 + 16x ds_store_b16
      int kk  = tid >> 2;         // 0..31
      int nn0 = (tid & 3) * 16;   // 0,16,32,48
      const unsigned short* gp = Bw + (size_t)(k0 + kk) * N + bn + nn0;
      Frag16 stg;
      stg.q[0] = *(const U4*)(gp);
      stg.q[1] = *(const U4*)(gp + 8);
#pragma unroll
      for (int i = 0; i < 16; ++i) Bt[nn0 + i][kk] = stg.s[i];
    }
    __syncthreads();

    // A fragments: lane = row (ln), K = 16*hi + h (contiguous)
    Frag16 a[2];
#pragma unroll
    for (int mt = 0; mt < 2; ++mt) {
      const unsigned short* ap =
          A + (size_t)(bm + wave * 32 + mt * 16 + ln) * K + k0;
      a[mt].q[0] = *(const U4*)(ap + 16 * hi);
      a[mt].q[1] = *(const U4*)(ap + 16 * hi + 8);
    }

#pragma unroll
    for (int nt = 0; nt < 4; ++nt) {
      // B fragment: lane = col (nt*16+ln), K = 16*hi + h -> contiguous in Bt row
      Frag16 b;
      const unsigned short* bp = &Bt[nt * 16 + ln][hi * 16];
      b.q[0] = *(const U4*)(bp);
      b.q[1] = *(const U4*)(bp + 8);
#pragma unroll
      for (int mt = 0; mt < 2; ++mt)
        acc[mt][nt] = WMMA_BF16(a[mt].v, b.v, acc[mt][nt]);
    }
  }

  // Epilogue. C layout: element r -> (row = r + 8*hi, col = lane&15) per tile.
#pragma unroll
  for (int mt = 0; mt < 2; ++mt) {
#pragma unroll
    for (int nt = 0; nt < 4; ++nt) {
#pragma unroll
      for (int r = 0; r < 8; ++r) {
        int gr = bm + wave * 32 + mt * 16 + r + 8 * hi;
        int gc = bn + nt * 16 + ln;
        float val = acc[mt][nt][r];
        if (op == OP_QUQV) {
          int l = gr >> 2, b = gr & 3;
          int h = gc >> 6, dh = gc & 63;
          size_t o = ((size_t)(b * H_ + h) * L_ + l) * DH_ + dh;
          o_qu[o] = f2bf(val + bias_u[h * DH_ + dh]);
          o_qv[o] = f2bf(val + bias_v[h * DH_ + dh]);
        } else if (op == OP_KV) {
          int l = gr >> 2, b = gr & 3;
          if (gc < HD_) {
            int h = gc >> 6, dh = gc & 63;
            o_k[((size_t)(b * H_ + h) * L_ + l) * DH_ + dh] = f2bf(val);
          } else {
            int c = gc - HD_;
            int h = c >> 6, dh = c & 63;
            // V stored transposed [b,h,dh,l] so PV B-fragments load contiguously
            o_vt[((size_t)(b * H_ + h) * DH_ + dh) * L_ + l] = f2bf(val);
          }
        } else if (op == OP_R) {
          int l = gr >> 2, b = gr & 3;
          int h = gc >> 6, dh = gc & 63;
          o_r[((size_t)(b * H_ + h) * L_ + l) * DH_ + dh] = f2bf(val);
        } else {  // OP_OUT: fp32 final output, row = l*B+b matches [L,B,D]
          o_f32[(size_t)gr * N + gc] = val;
        }
      }
    }
  }
}

// ---------------------------------------------------------------------------
// Flash attention with Transformer-XL relative shift.
// Grid: (L/64, B*H). 128 threads = 4 waves; each wave owns 16 query rows and
// streams key tiles of 32 causally (j0 <= qi0 always for 32-aligned tiles).
// score[i,j] = (qu_i . k_j + qv_i . r[L-1-i+j]) * SCALE  for j <= i.
// BD handled via band GEMM G = QV @ Rband^T then diagonal gather from LDS.
// ---------------------------------------------------------------------------
__global__ __launch_bounds__(128)
void flash_relattn(const unsigned short* __restrict__ QU,
                   const unsigned short* __restrict__ QV,
                   const unsigned short* __restrict__ Kc,
                   const unsigned short* __restrict__ Vt,
                   const unsigned short* __restrict__ Rc,
                   unsigned short* __restrict__ attn) {
  __shared__ float Gs[4][16][49];
  __shared__ unsigned short Ps[4][16][32];
  const int tid  = threadIdx.x;
  const int wave = tid >> 5, lane = tid & 31;
  const int hi   = lane >> 4, ln = lane & 15;
  const int bh = blockIdx.y;          // b*H + h
  const int b = bh >> 4, h = bh & 15;
  const int qi0 = blockIdx.x * 64 + wave * 16;

  const unsigned short* QUb = QU + ((size_t)bh * L_ + qi0) * DH_;
  const unsigned short* QVb = QV + ((size_t)bh * L_ + qi0) * DH_;
  const unsigned short* Kb  = Kc + (size_t)bh * L_ * DH_;
  const unsigned short* Vb  = Vt + (size_t)bh * DH_ * L_;
  const unsigned short* Rb  = Rc + (size_t)bh * L_ * DH_;

  // Persistent A-fragments for QU/QV (16 rows x 64 K, two K-steps of 32)
  Frag16 au[2], av[2];
  {
    const unsigned short* qur = QUb + (size_t)ln * DH_;
    const unsigned short* qvr = QVb + (size_t)ln * DH_;
#pragma unroll
    for (int s = 0; s < 2; ++s) {
      au[s].q[0] = *(const U4*)(qur + 32 * s + 16 * hi);
      au[s].q[1] = *(const U4*)(qur + 32 * s + 16 * hi + 8);
      av[s].q[0] = *(const U4*)(qvr + 32 * s + 16 * hi);
      av[s].q[1] = *(const U4*)(qvr + 32 * s + 16 * hi + 8);
    }
  }

  float mrow[8], ssum[8];
  v8f o[4];
#pragma unroll
  for (int r = 0; r < 8; ++r) { mrow[r] = -3.0e38f; ssum[r] = 0.f; }
#pragma unroll
  for (int t = 0; t < 4; ++t) o[t] = v8f_zero();

  for (int j0 = 0; j0 <= qi0 + 15; j0 += 32) {
    // ---- AC = QU @ K^T : two 16x16 score frags over 32 keys
    v8f c0 = v8f_zero(), c1 = v8f_zero();
#pragma unroll
    for (int s = 0; s < 2; ++s) {
      Frag16 kb0, kb1;  // B-frag: lane = key col, K(=dh) = s*32 + 16*hi + h
      const unsigned short* kp0 = Kb + (size_t)(j0 + ln) * DH_ + 32 * s + 16 * hi;
      const unsigned short* kp1 = Kb + (size_t)(j0 + 16 + ln) * DH_ + 32 * s + 16 * hi;
      kb0.q[0] = *(const U4*)(kp0); kb0.q[1] = *(const U4*)(kp0 + 8);
      kb1.q[0] = *(const U4*)(kp1); kb1.q[1] = *(const U4*)(kp1 + 8);
      c0 = WMMA_BF16(au[s].v, kb0.v, c0);
      c1 = WMMA_BF16(au[s].v, kb1.v, c1);
    }

    // ---- G = QV @ Rband^T over 48-row band; BD[ii,jj] = G[ii, jj-ii+15]
    const int base = (L_ - 1) - qi0 + j0;
#pragma unroll
    for (int nt = 0; nt < 3; ++nt) {
      v8f g = v8f_zero();
      int m = base - 15 + nt * 16 + ln;
#pragma unroll
      for (int s = 0; s < 2; ++s) {
        Frag16 rb;
        if (m >= 0 && m < L_) {
          const unsigned short* rp = Rb + (size_t)m * DH_ + 32 * s + 16 * hi;
          rb.q[0] = *(const U4*)(rp); rb.q[1] = *(const U4*)(rp + 8);
        } else {
          rb.q[0] = U4{0u, 0u, 0u, 0u}; rb.q[1] = U4{0u, 0u, 0u, 0u};
        }
        g = WMMA_BF16(av[s].v, rb.v, g);
      }
#pragma unroll
      for (int r = 0; r < 8; ++r)
        Gs[wave][r + 8 * hi][nt * 16 + ln] = g[r];
    }

    // ---- assemble scores (AC + shifted BD), causal mask, scale
    float sc[2][8];
#pragma unroll
    for (int nt = 0; nt < 2; ++nt) {
      v8f cc = nt ? c1 : c0;
#pragma unroll
      for (int r = 0; r < 8; ++r) {
        int ii = r + 8 * hi;
        int jj = nt * 16 + ln;
        float v = (cc[r] + Gs[wave][ii][jj - ii + 15]) * 0.125f;  // SCALE=1/8
        if (j0 + jj > qi0 + ii) v = -1.0e30f;
        sc[nt][r] = v;
      }
    }

    // ---- online softmax (row reductions across 16-lane halves)
#pragma unroll
    for (int r = 0; r < 8; ++r) {
      float rm = fmaxf(sc[0][r], sc[1][r]);
#pragma unroll
      for (int msk = 1; msk < 16; msk <<= 1)
        rm = fmaxf(rm, __shfl_xor(rm, msk, 32));
      float mn  = fmaxf(mrow[r], rm);
      float scl = __expf(mrow[r] - mn);
      float p0  = __expf(sc[0][r] - mn);
      float p1  = __expf(sc[1][r] - mn);
      float rs  = p0 + p1;
#pragma unroll
      for (int msk = 1; msk < 16; msk <<= 1)
        rs += __shfl_xor(rs, msk, 32);
      ssum[r] = ssum[r] * scl + rs;
      mrow[r] = mn;
#pragma unroll
      for (int t = 0; t < 4; ++t) o[t][r] *= scl;
      int ii = r + 8 * hi;
      Ps[wave][ii][ln]      = f2bf(p0);
      Ps[wave][ii][16 + ln] = f2bf(p1);
    }

    // ---- P as A-fragment (16x32), LDS ops are in-order within a wave
    Frag16 pf;
    const unsigned short* pp = &Ps[wave][ln][0];
    pf.q[0] = *(const U4*)(pp + 16 * hi);
    pf.q[1] = *(const U4*)(pp + 16 * hi + 8);

    // ---- O += P @ V  (V transposed: contiguous key runs per fragment)
#pragma unroll
    for (int nt = 0; nt < 4; ++nt) {
      Frag16 vf;
      const unsigned short* vp = Vb + (size_t)(nt * 16 + ln) * L_ + j0 + 16 * hi;
      vf.q[0] = *(const U4*)(vp); vf.q[1] = *(const U4*)(vp + 8);
      o[nt] = WMMA_BF16(pf.v, vf.v, o[nt]);
    }
  }

  // ---- normalize and store attn_vec as bf16 [L*B, H*DH]
#pragma unroll
  for (int r = 0; r < 8; ++r) {
    float inv = 1.0f / ssum[r];
    int gl = qi0 + r + 8 * hi;
    size_t row = (size_t)(gl * B_ + b) * HD_;
#pragma unroll
    for (int nt = 0; nt < 4; ++nt)
      attn[row + h * DH_ + nt * 16 + ln] = f2bf(o[nt][r] * inv);
  }
}

// ---------------------------------------------------------------------------
extern "C" void kernel_launch(void* const* d_in, const int* in_sizes, int n_in,
                              void* d_out, int out_size, void* d_ws, size_t ws_size,
                              hipStream_t stream) {
  const float* x    = (const float*)d_in[0];
  const float* pos  = (const float*)d_in[1];
  const float* bu   = (const float*)d_in[2];
  const float* bv   = (const float*)d_in[3];
  // d_in[4] = causal mask (recomputed analytically in-kernel)
  const float* Wq   = (const float*)d_in[5];
  const float* Wkv  = (const float*)d_in[6];
  const float* Wrel = (const float*)d_in[7];
  const float* Wo   = (const float*)d_in[8];
  float* out = (float*)d_out;

  unsigned short* w = (unsigned short*)d_ws;
  const size_t nXB = (size_t)LB_ * D_;                 // 8M elems
  const size_t nBH = (size_t)B_ * H_ * L_ * DH_;       // 8M elems
  unsigned short* Xb    = w; w += nXB;
  unsigned short* Pb    = w; w += nXB;
  unsigned short* Wqb   = w; w += (size_t)D_ * HD_;
  unsigned short* Wkvb  = w; w += (size_t)D_ * 2 * HD_;
  unsigned short* Wrelb = w; w += (size_t)D_ * HD_;
  unsigned short* Wob   = w; w += (size_t)HD_ * D_;
  unsigned short* QUw   = w; w += nBH;
  unsigned short* QVw   = w; w += nBH;
  unsigned short* Kw    = w; w += nBH;
  unsigned short* Vtw   = w; w += nBH;
  unsigned short* Rw    = w; w += nBH;
  unsigned short* Aw    = w; w += nXB;

  auto cvt = [&](const float* s, unsigned short* d, size_t n) {
    cvt_f32_bf16<<<dim3((unsigned)((n + 255) / 256)), dim3(256), 0, stream>>>(s, d, (int)n);
  };
  cvt(x,    Xb,    nXB);
  cvt(pos,  Pb,    nXB);
  cvt(Wq,   Wqb,   (size_t)D_ * HD_);
  cvt(Wkv,  Wkvb,  (size_t)D_ * 2 * HD_);
  cvt(Wrel, Wrelb, (size_t)D_ * HD_);
  cvt(Wo,   Wob,   (size_t)HD_ * D_);

  dim3 blk(128);
  // Q projection -> QU, QV (bias add fused)
  gemm_wmma<<<dim3(LB_ / 128, HD_ / 64), blk, 0, stream>>>(
      Xb, Wqb, LB_, HD_, D_, OP_QUQV, bu, bv,
      QUw, QVw, nullptr, nullptr, nullptr, nullptr);
  // KV projection -> K [b,h,l,dh], V transposed [b,h,dh,l]
  gemm_wmma<<<dim3(LB_ / 128, (2 * HD_) / 64), blk, 0, stream>>>(
      Xb, Wkvb, LB_, 2 * HD_, D_, OP_KV, nullptr, nullptr,
      nullptr, nullptr, Kw, Vtw, nullptr, nullptr);
  // R projection -> R [b,h,l,dh]
  gemm_wmma<<<dim3(LB_ / 128, HD_ / 64), blk, 0, stream>>>(
      Pb, Wrelb, LB_, HD_, D_, OP_R, nullptr, nullptr,
      nullptr, nullptr, nullptr, nullptr, Rw, nullptr);
  // Flash attention with rel-shift
  flash_relattn<<<dim3(L_ / 64, B_ * H_), blk, 0, stream>>>(
      QUw, QVw, Kw, Vtw, Rw, Aw);
  // Output projection -> fp32 out
  gemm_wmma<<<dim3(LB_ / 128, D_ / 64), blk, 0, stream>>>(
      Aw, Wob, LB_, D_, HD_, OP_OUT, nullptr, nullptr,
      nullptr, nullptr, nullptr, nullptr, nullptr, out);
}
